// TemporalAttention_7473243095531
// MI455X (gfx1250) — matrix operand order repaired and checked by reading, not compiled, not run on hardware
//
#include <hip/hip_runtime.h>

typedef __attribute__((ext_vector_type(16))) _Float16 v16h;
typedef __attribute__((ext_vector_type(8)))  _Float16 v8h;
typedef __attribute__((ext_vector_type(8)))  float    v8f;

#define B_  32
#define C_  256
#define L_  1024
#define QK_ 256
#define V_  256

// padded LDS row lengths (halves): keep 16B alignment, spread banks
#define QLD 264   // 528B rows (33*16B)
#define VLD 40    // 80B rows (5*16B)
#define PLD 40
#define ALD 264

// ---------------------------------------------------------------------------
// WMMA helpers (wave32, v_wmma_f32_16x16x32_f16)
// ---------------------------------------------------------------------------
__device__ __forceinline__ v8f vzero8() {
  v8f z;
#pragma unroll
  for (int r = 0; r < 8; ++r) z[r] = 0.0f;
  return z;
}

__device__ __forceinline__ v8f wmma_ff(v16h a, v16h b, v8f c) {
  return __builtin_amdgcn_wmma_f32_16x16x32_f16(
      false, a, false, b, (short)0, c, false, false);
}

// A fragment: 16x32 (MxK) fp16 from row-major [rows, ld].
// lanes 0-15 -> M=lane, e0..7=K0..7, e8..15=K16..23
// lanes16-31 -> M=lane-16, e0..7=K8..15, e8..15=K24..31
__device__ __forceinline__ v16h load_frag_a(const _Float16* base, int row0,
                                            int k0, int ld) {
  int lane = threadIdx.x & 31;
  int hi   = lane >> 4;
  int m    = lane & 15;
  const _Float16* p = base + (size_t)(row0 + m) * ld + k0 + 8 * hi;
  v8h lo = *(const v8h*)(p);
  v8h hv = *(const v8h*)(p + 16);
  v16h a;
#pragma unroll
  for (int e = 0; e < 8; ++e) { a[e] = lo[e]; a[8 + e] = hv[e]; }
  return a;
}

// B fragment: 32x16 (KxN) fp16 where B[k][n] = W[n0+n][k0+k], W row-major [*, ld].
__device__ __forceinline__ v16h load_frag_b(const _Float16* w, int n0,
                                            int k0, int ld) {
  int lane = threadIdx.x & 31;
  int hi   = lane >> 4;
  int n    = lane & 15;
  const _Float16* p = w + (size_t)(n0 + n) * ld + k0 + 8 * hi;
  v8h lo = *(const v8h*)(p);
  v8h hv = *(const v8h*)(p + 16);
  v16h b;
#pragma unroll
  for (int e = 0; e < 8; ++e) { b[e] = lo[e]; b[8 + e] = hv[e]; }
  return b;
}

__device__ __forceinline__ float rmax16(float v) {
#pragma unroll
  for (int o = 1; o < 16; o <<= 1) v = fmaxf(v, __shfl_xor(v, o, 16));
  return v;
}
__device__ __forceinline__ float rsum16(float v) {
#pragma unroll
  for (int o = 1; o < 16; o <<= 1) v += __shfl_xor(v, o, 16);
  return v;
}

// CDNA5 async global->LDS copy (ASYNCcnt-tracked, no VGPR data path)
__device__ __forceinline__ void async_copy_b128(unsigned lds_byte_addr,
                                                const void* gptr) {
  asm volatile("global_load_async_to_lds_b128 %0, %1, off"
               :: "v"(lds_byte_addr), "v"(gptr) : "memory");
}
__device__ __forceinline__ void wait_async0() {
  asm volatile("s_wait_asynccnt 0" ::: "memory");
}
__device__ __forceinline__ void wait_ds0() {
  asm volatile("s_wait_dscnt 0" ::: "memory");
}

// ---------------------------------------------------------------------------
// Kernel 0: transpose + convert x [B,C,L] f32 -> xT [B,L,C] f16 (LDS tiled)
// ---------------------------------------------------------------------------
__global__ void __launch_bounds__(256)
k_transpose(const float* __restrict__ x, _Float16* __restrict__ xT) {
  __shared__ float tile[32][33];
  int b  = blockIdx.z;
  int c0 = blockIdx.y * 32;
  int l0 = blockIdx.x * 32;
  int tx = threadIdx.x;
  int ty = threadIdx.y;
  const float* xb = x + (size_t)b * C_ * L_;
#pragma unroll
  for (int i = 0; i < 4; ++i)
    tile[ty + i * 8][tx] = xb[(size_t)(c0 + ty + i * 8) * L_ + l0 + tx];
  __syncthreads();
  _Float16* xTb = xT + (size_t)b * L_ * C_;
#pragma unroll
  for (int i = 0; i < 4; ++i)
    xTb[(size_t)(l0 + ty + i * 8) * C_ + c0 + tx] = (_Float16)tile[tx][ty + i * 8];
}

// ---------------------------------------------------------------------------
// Kernel 0b: convert weights to fp16 (all 256x256)
// ---------------------------------------------------------------------------
__global__ void __launch_bounds__(256)
k_cvt_w(const float* __restrict__ Wk, const float* __restrict__ Wq,
        const float* __restrict__ Wv, const float* __restrict__ Wo,
        _Float16* __restrict__ Wkh, _Float16* __restrict__ Wqh,
        _Float16* __restrict__ Wvh, _Float16* __restrict__ Woh) {
  int i = blockIdx.x * blockDim.x + threadIdx.x;
  if (i < QK_ * C_) {
    Wkh[i] = (_Float16)Wk[i];
    Wqh[i] = (_Float16)Wq[i];
    Wvh[i] = (_Float16)Wv[i];
    Woh[i] = (_Float16)Wo[i];
  }
}

// ---------------------------------------------------------------------------
// Kernel 1: fused K/Q/V projections (unchanged layout strategy):
//   key[l,o], qT[l,o] (=query^T), vT[o,l] (=value^T), all fp16
// ---------------------------------------------------------------------------
__global__ void __launch_bounds__(256)
k_proj(const _Float16* __restrict__ xT,
       const _Float16* __restrict__ Wkh, const float* __restrict__ bk,
       const _Float16* __restrict__ Wqh, const float* __restrict__ bq,
       const _Float16* __restrict__ Wvh, const float* __restrict__ bv,
       _Float16* __restrict__ keyh, _Float16* __restrict__ qTh,
       _Float16* __restrict__ vTh) {
  int b    = blockIdx.y;
  int wave = threadIdx.x >> 5;
  int lane = threadIdx.x & 31;
  int hi   = lane >> 4;
  int nl   = lane & 15;
  int row0 = blockIdx.x * 128 + wave * 16;

  const _Float16* xTb = xT + (size_t)b * L_ * C_;
  v16h afr[8];
#pragma unroll
  for (int ks = 0; ks < 8; ++ks) afr[ks] = load_frag_a(xTb, row0, ks * 32, C_);

  _Float16* keyb = keyh + (size_t)b * L_ * QK_;
  _Float16* qTb  = qTh  + (size_t)b * L_ * QK_;
  _Float16* vTb  = vTh  + (size_t)b * V_ * L_;

#pragma unroll 1
  for (int nt = 0; nt < 16; ++nt) {
    int n0 = nt * 16;
    v8f ck = vzero8(), cq = vzero8(), cv = vzero8();
#pragma unroll
    for (int ks = 0; ks < 8; ++ks) {
      ck = wmma_ff(afr[ks], load_frag_b(Wkh, n0, ks * 32, C_), ck);
      cq = wmma_ff(afr[ks], load_frag_b(Wqh, n0, ks * 32, C_), cq);
      cv = wmma_ff(afr[ks], load_frag_b(Wvh, n0, ks * 32, C_), cv);
    }
    float bkk = bk[n0 + nl], bqq = bq[n0 + nl], bvv = bv[n0 + nl];
#pragma unroll
    for (int r = 0; r < 8; ++r) {
      int l = row0 + r + 8 * hi;
      keyb[(size_t)l * QK_ + n0 + nl]  = (_Float16)(ck[r] + bkk);
      qTb [(size_t)l * QK_ + n0 + nl]  = (_Float16)(cq[r] + bqq);
      vTb [(size_t)(n0 + nl) * L_ + l] = (_Float16)(cv[r] + bvv);
    }
  }
}

// ---------------------------------------------------------------------------
// Kernel 2: fused flash attention + output projection.
// 4 waves/block, 16 query rows per wave.  qT/vT j-chunks (32 wide) are staged
// once per block into double-buffered LDS via global_load_async_to_lds_b128,
// overlapped with the previous chunk's WMMAs (ASYNCcnt + barrier sync).
// ---------------------------------------------------------------------------
__global__ void __launch_bounds__(128)
k_attn(const _Float16* __restrict__ keyh, const _Float16* __restrict__ qTh,
       const _Float16* __restrict__ vTh,  const _Float16* __restrict__ Woh,
       const float* __restrict__ bo, const int* __restrict__ maskp,
       float* __restrict__ out) {
  __shared__ __align__(16) _Float16 Qs[2][32][QLD];    // qT rows j0..j0+31
  __shared__ __align__(16) _Float16 Vs[2][256][VLD];   // vT cols j0..j0+31
  __shared__ __align__(16) _Float16 Plds[4][16][PLD];  // P tile D->A relayout
  __shared__ __align__(16) _Float16 AVlds[4][16][ALD]; // AV D->B relayout

  int b    = blockIdx.y;
  int tid  = threadIdx.x;
  int wave = tid >> 5;
  int lane = tid & 31;
  int hi   = lane >> 4;
  int nl   = lane & 15;
  int i0   = blockIdx.x * 64 + wave * 16;
  int msk  = maskp[0];

  const _Float16* keyb = keyh + (size_t)b * L_ * QK_;
  const _Float16* qTb  = qTh  + (size_t)b * L_ * QK_;
  const _Float16* vTb  = vTh  + (size_t)b * V_ * L_;

  // cooperative async stage of one j-chunk into buffer `buf`
  auto stage_chunk = [&](int jc, int buf) {
    int j0 = jc * 32;
    // qT: 32 rows x 512B  (1024 x 16B segments, 8 per thread)
    unsigned qbase = (unsigned)(size_t)&Qs[buf][0][0];
    const _Float16* qsrc = qTb + (size_t)j0 * QK_;
#pragma unroll
    for (int it = 0; it < 8; ++it) {
      int seg = tid + it * 128;
      int row = seg >> 5, col = (seg & 31) * 8;  // halves
      async_copy_b128(qbase + (unsigned)(row * QLD + col) * 2,
                      qsrc + (size_t)row * QK_ + col);
    }
    // vT: 256 rows x 64B  (1024 x 16B segments, 8 per thread)
    unsigned vbase = (unsigned)(size_t)&Vs[buf][0][0];
    const _Float16* vsrc = vTb + j0;
#pragma unroll
    for (int it = 0; it < 8; ++it) {
      int seg = tid + it * 128;
      int row = seg >> 2, col = (seg & 3) * 8;   // halves
      async_copy_b128(vbase + (unsigned)(row * VLD + col) * 2,
                      vsrc + (size_t)row * L_ + col);
    }
  };

  // K row fragments for this wave's 16 rows (reused for every j chunk)
  v16h ka[8];
#pragma unroll
  for (int ks = 0; ks < 8; ++ks) ka[ks] = load_frag_a(keyb, i0, ks * 32, QK_);

  float Mrun[8], Lrun[8];
  v8f O[16];
#pragma unroll
  for (int r = 0; r < 8; ++r) { Mrun[r] = -1e30f; Lrun[r] = 0.0f; }
#pragma unroll
  for (int t = 0; t < 16; ++t) O[t] = vzero8();

  _Float16 (*Pw)[PLD] = Plds[wave];

  // preload chunk 0
  stage_chunk(0, 0);
  wait_async0();
  __syncthreads();

#pragma unroll 1
  for (int jc = 0; jc < 32; ++jc) {
    int cur = jc & 1, nxt = cur ^ 1;
    int j0 = jc * 32;
    if (jc + 1 < 32) stage_chunk(jc + 1, nxt);  // DMA next chunk during compute

    const _Float16* Qc = &Qs[cur][0][0];
    const _Float16* Vc = &Vs[cur][0][0];

    // ---- S tiles: rows i0..i0+15, cols j0..j0+31, K=QK=256 (B frags from LDS) ----
    v8f s0 = vzero8(), s1 = vzero8();
#pragma unroll
    for (int ks = 0; ks < 8; ++ks) {
      s0 = wmma_ff(ka[ks], load_frag_b(Qc, 0,  ks * 32, QLD), s0);
      s1 = wmma_ff(ka[ks], load_frag_b(Qc, 16, ks * 32, QLD), s1);
    }
    // ---- multiplicative triu mask + online softmax ----
    float alph[8];
#pragma unroll
    for (int r = 0; r < 8; ++r) {
      int ii = i0 + r + 8 * hi;
      if (msk && (j0 + nl)      < ii) s0[r] = 0.0f;
      if (msk && (j0 + 16 + nl) < ii) s1[r] = 0.0f;
      float cmax = rmax16(fmaxf(s0[r], s1[r]));
      float Mn   = fmaxf(Mrun[r], cmax);
      float a    = __expf(Mrun[r] - Mn);
      Mrun[r]    = Mn;
      float p0 = __expf(s0[r] - Mn);
      float p1 = __expf(s1[r] - Mn);
      Lrun[r] = Lrun[r] * a + rsum16(p0 + p1);
      alph[r] = a;
      Pw[r + 8 * hi][nl]      = (_Float16)p0;
      Pw[r + 8 * hi][16 + nl] = (_Float16)p1;
    }
#pragma unroll
    for (int t = 0; t < 16; ++t)
#pragma unroll
      for (int r = 0; r < 8; ++r) O[t][r] *= alph[r];

    wait_ds0();
    __builtin_amdgcn_wave_barrier();

    // ---- O += P (16x32) x V-chunk (32x256), B frags from LDS ----
    v16h pa = load_frag_a(&Pw[0][0], 0, 0, PLD);
#pragma unroll
    for (int vt = 0; vt < 16; ++vt)
      O[vt] = wmma_ff(pa, load_frag_b(Vc, vt * 16, 0, VLD), O[vt]);

    wait_async0();     // next chunk's DMA landed
    __syncthreads();   // all waves done reading cur, data of nxt visible
  }

  // ---- normalize, stage AV (16 x 256) into LDS for the fused Wo GEMM ----
  _Float16 (*AVw)[ALD] = AVlds[wave];
#pragma unroll
  for (int vt = 0; vt < 16; ++vt)
#pragma unroll
    for (int r = 0; r < 8; ++r)
      AVw[r + 8 * hi][vt * 16 + nl] = (_Float16)(O[vt][r] / Lrun[r]);

  wait_ds0();
  __builtin_amdgcn_wave_barrier();

  // ---- out[c,l] = sum_v Wo[c,v] * AV[l,v] + bo[c] ----
  float* outb = out + (size_t)b * C_ * L_;
#pragma unroll 1
  for (int ct = 0; ct < 16; ++ct) {
    v8f oc = vzero8();
#pragma unroll
    for (int ks = 0; ks < 8; ++ks)
      oc = wmma_ff(load_frag_a(Woh, ct * 16, ks * 32, V_),
                   load_frag_b(&AVw[0][0], 0, ks * 32, ALD), oc);
#pragma unroll
    for (int r = 0; r < 8; ++r) {
      int c = ct * 16 + r + 8 * hi;
      int l = i0 + nl;
      outb[(size_t)c * L_ + l] = oc[r] + bo[c];
    }
  }
}

// ---------------------------------------------------------------------------
// Host launch
// ---------------------------------------------------------------------------
extern "C" void kernel_launch(void* const* d_in, const int* in_sizes, int n_in,
                              void* d_out, int out_size, void* d_ws, size_t ws_size,
                              hipStream_t stream) {
  const float* x  = (const float*)d_in[0];
  const float* Wk = (const float*)d_in[1];
  const float* bk = (const float*)d_in[2];
  const float* Wq = (const float*)d_in[3];
  const float* bq = (const float*)d_in[4];
  const float* Wv = (const float*)d_in[5];
  const float* bv = (const float*)d_in[6];
  const float* Wo = (const float*)d_in[7];
  const float* bo = (const float*)d_in[8];
  const int*   mk = (const int*)d_in[9];
  float* out = (float*)d_out;

  const size_t nBLC = (size_t)B_ * L_ * C_;
  const size_t nW   = (size_t)QK_ * C_;
  _Float16* ws = (_Float16*)d_ws;
  _Float16* xT   = ws;
  _Float16* keyh = xT   + nBLC;
  _Float16* qTh  = keyh + nBLC;
  _Float16* vTh  = qTh  + nBLC;
  _Float16* Wkh  = vTh  + nBLC;
  _Float16* Wqh  = Wkh  + nW;
  _Float16* Wvh  = Wqh  + nW;
  _Float16* Woh  = Wvh  + nW;

  dim3 g0(L_ / 32, C_ / 32, B_), b0(32, 8);
  k_transpose<<<g0, b0, 0, stream>>>(x, xT);

  k_cvt_w<<<(QK_ * C_ + 255) / 256, 256, 0, stream>>>(Wk, Wq, Wv, Wo,
                                                      Wkh, Wqh, Wvh, Woh);

  k_proj<<<dim3(L_ / 128, B_), 256, 0, stream>>>(xT, Wkh, bk, Wqh, bq, Wvh, bv,
                                                 keyh, qTh, vTh);

  k_attn<<<dim3(L_ / 64, B_), 128, 0, stream>>>(keyh, qTh, vTh, Woh, bo, mk, out);
}
